// DSNet_65687229826150
// MI455X (gfx1250) — matrix operand order repaired and checked by reading, not compiled, not run on hardware
//
#include <hip/hip_runtime.h>
#include <hip/hip_bf16.h>

typedef _Float16 half_t;
typedef __attribute__((ext_vector_type(16))) _Float16 v16h;
typedef __attribute__((ext_vector_type(8)))  _Float16 v8h;
typedef __attribute__((ext_vector_type(8)))  float    v8f;

#define BB    8
#define NPTS  2048
#define FRED  64
#define NFEA  128
#define KG    8
#define KLU   16
#define NKG   (NPTS*KG)          // 16384 grouped positions per batch
#define BN_EPS 1e-5f

__device__ __forceinline__ v16h cat8(v8h a, v8h b) {
  return __builtin_shufflevector(a, b, 0,1,2,3,4,5,6,7,8,9,10,11,12,13,14,15);
}

// ---------------------------------------------------------------------------
// kNN: per point, top-16 nearest (2-D coords, includes self), sorted ascending.
// ---------------------------------------------------------------------------
__global__ void knn_kernel(const float* __restrict__ xyz, int* __restrict__ idx16) {
  __shared__ float sx[NPTS];
  __shared__ float sy[NPTS];
  const int b = blockIdx.y;
  const float* xb = xyz + (size_t)b * 3 * NPTS;
  for (int i = threadIdx.x; i < NPTS; i += blockDim.x) { sx[i] = xb[i]; sy[i] = xb[NPTS + i]; }
  __syncthreads();
  const int n = blockIdx.x * blockDim.x + threadIdx.x;
  const float px = sx[n], py = sy[n];
  float bd[KLU]; int bi[KLU];
#pragma unroll
  for (int i = 0; i < KLU; ++i) { bd[i] = 3.4e38f; bi[i] = 0; }
  for (int m = 0; m < NPTS; ++m) {
    const float dx = px - sx[m], dy = py - sy[m];
    float cd = dx * dx + dy * dy;
    int ci = m;
    if (cd < bd[KLU - 1]) {
#pragma unroll
      for (int j = 0; j < KLU; ++j) {
        if (cd < bd[j]) { float td = bd[j]; int ti = bi[j]; bd[j] = cd; bi[j] = ci; cd = td; ci = ti; }
      }
    }
  }
  int* out = idx16 + ((size_t)b * NPTS + n) * KLU;
#pragma unroll
  for (int j = 0; j < KLU; ++j) out[j] = bi[j];
}

// ---------------------------------------------------------------------------
// Weight cast f32 -> f16 (row-major kept: A-operand layout is two contiguous
// 8-half runs per lane -> b128 loads)
// ---------------------------------------------------------------------------
__global__ void cast_f16_kernel(const float* __restrict__ in, half_t* __restrict__ out, int n) {
  int i = blockIdx.x * blockDim.x + threadIdx.x;
  if (i < n) out[i] = (half_t)in[i];
}

// feat [B][NFEA][N] f32 -> featT [B][N][NFEA] f16  (B-operand, [L][K] layout)
__global__ void cast_feat_t_kernel(const float* __restrict__ in, half_t* __restrict__ out) {
  const size_t i = (size_t)blockIdx.x * blockDim.x + threadIdx.x;
  if (i >= (size_t)BB * NFEA * NPTS) return;
  const int n = (int)(i % NPTS);
  const int c = (int)((i / NPTS) % NFEA);
  const int b = (int)(i / ((size_t)NFEA * NPTS));
  out[((size_t)b * NPTS + n) * NFEA + c] = (half_t)in[i];
}

// ---------------------------------------------------------------------------
// WMMA GEMM, per batch:  Y[b][M][L] = W[M,K] @ Xt[b][L][K]^T + bias
// One wave -> 16(M) x 64(L) tile; A fragment reused across 4 WMMAs.
// Fragments (C per ISA 7.12.2; A/B share contiguous k order, not the ISA table order):
//   A: row = lane&15; K = hi*16 + j, contiguous -> 2 x b128
//   B: col = lane&15; K = hi*16 + j, contiguous in [L][K] layout -> 2 x b128
//   C: c[r] -> M = r + 8*hi, N = lane&15
// ---------------------------------------------------------------------------
__global__ void gemm_bias_f16(const half_t* __restrict__ W, const half_t* __restrict__ Xt,
                              const float* __restrict__ bias, float* __restrict__ Y,
                              int M, int K, int L) {
  const int lane = threadIdx.x;
  const int hi = lane >> 4, lo = lane & 15;
  const int mrow = blockIdx.y * 16;
  const int b    = blockIdx.z;
  const int col0 = blockIdx.x * 64 + lo;
  const half_t* arow = W + (size_t)(mrow + lo) * K;
  const half_t* Xb   = Xt + (size_t)b * L * K;
  const half_t* br0 = Xb + (size_t)(col0     ) * K;
  const half_t* br1 = Xb + (size_t)(col0 + 16) * K;
  const half_t* br2 = Xb + (size_t)(col0 + 32) * K;
  const half_t* br3 = Xb + (size_t)(col0 + 48) * K;
  v8f a0 = {}, a1 = {}, a2 = {}, a3 = {};
  for (int kb = 0; kb < K; kb += 32) {
    __builtin_prefetch(arow + kb + 64, 0, 1);
    const v16h a  = cat8(*(const v8h*)(arow + kb + hi * 16),
                         *(const v8h*)(arow + kb + hi * 16 + 8));
    const v16h b0 = cat8(*(const v8h*)(br0 + kb + hi * 16),
                         *(const v8h*)(br0 + kb + hi * 16 + 8));
    const v16h b1 = cat8(*(const v8h*)(br1 + kb + hi * 16),
                         *(const v8h*)(br1 + kb + hi * 16 + 8));
    const v16h b2 = cat8(*(const v8h*)(br2 + kb + hi * 16),
                         *(const v8h*)(br2 + kb + hi * 16 + 8));
    const v16h b3 = cat8(*(const v8h*)(br3 + kb + hi * 16),
                         *(const v8h*)(br3 + kb + hi * 16 + 8));
    a0 = __builtin_amdgcn_wmma_f32_16x16x32_f16(false, a, false, b0, (short)0, a0, false, false);
    a1 = __builtin_amdgcn_wmma_f32_16x16x32_f16(false, a, false, b1, (short)0, a1, false, false);
    a2 = __builtin_amdgcn_wmma_f32_16x16x32_f16(false, a, false, b2, (short)0, a2, false, false);
    a3 = __builtin_amdgcn_wmma_f32_16x16x32_f16(false, a, false, b3, (short)0, a3, false, false);
  }
  float* Yb = Y + (size_t)b * M * L;
#pragma unroll
  for (int r = 0; r < 8; ++r) {
    const int m = mrow + r + 8 * hi;
    const float bs = bias[m];
    float* ym = Yb + (size_t)m * L;
    ym[col0]      = a0[r] + bs;
    ym[col0 + 16] = a1[r] + bs;
    ym[col0 + 32] = a2[r] + bs;
    ym[col0 + 48] = a3[r] + bs;
  }
}

// ---------------------------------------------------------------------------
// Group-MLP WMMA GEMM with faithful torch-.view gather:
//   H[b][o][p] = sum_f Wf[o,f] * x1[b, f, idx[b,(k*N+n)/8,(k*N+n)%8]] + bf[o]
// p = n*KG + k. x1 stored transposed [B][N][FRED] f16, so the gathered
// neighbor is one contiguous 64-half row -> pure b128 fragment feeds.
// ---------------------------------------------------------------------------
__global__ void gemm_group_f16(const half_t* __restrict__ Wf, const half_t* __restrict__ X1t,
                               const int* __restrict__ idx16, const float* __restrict__ bias,
                               float* __restrict__ H) {
  const int lane = threadIdx.x;
  const int hi = lane >> 4, lo = lane & 15;
  const int mrow = blockIdx.y * 16;
  const int b    = blockIdx.z;
  const int p0   = blockIdx.x * 64 + lo;
  const half_t* arow = Wf + (size_t)(mrow + lo) * FRED;
  const half_t* Xb   = X1t + (size_t)b * NPTS * FRED;
  const int* ib = idx16 + (size_t)b * NPTS * KLU;
  const half_t* br[4];
#pragma unroll
  for (int q = 0; q < 4; ++q) {
    const int p = p0 + q * 16;
    const int n = p >> 3, k = p & 7;
    const int qq = k * NPTS + n;                    // torch .view scramble
    const int src = ib[(qq >> 3) * KLU + (qq & 7)];
    br[q] = Xb + (size_t)src * FRED;
  }
  v8f a0 = {}, a1 = {}, a2 = {}, a3 = {};
  for (int kb = 0; kb < FRED; kb += 32) {
    const v16h a  = cat8(*(const v8h*)(arow + kb + hi * 16),
                         *(const v8h*)(arow + kb + hi * 16 + 8));
    const v16h b0 = cat8(*(const v8h*)(br[0] + kb + hi * 16),
                         *(const v8h*)(br[0] + kb + hi * 16 + 8));
    const v16h b1 = cat8(*(const v8h*)(br[1] + kb + hi * 16),
                         *(const v8h*)(br[1] + kb + hi * 16 + 8));
    const v16h b2 = cat8(*(const v8h*)(br[2] + kb + hi * 16),
                         *(const v8h*)(br[2] + kb + hi * 16 + 8));
    const v16h b3 = cat8(*(const v8h*)(br[3] + kb + hi * 16),
                         *(const v8h*)(br[3] + kb + hi * 16 + 8));
    a0 = __builtin_amdgcn_wmma_f32_16x16x32_f16(false, a, false, b0, (short)0, a0, false, false);
    a1 = __builtin_amdgcn_wmma_f32_16x16x32_f16(false, a, false, b1, (short)0, a1, false, false);
    a2 = __builtin_amdgcn_wmma_f32_16x16x32_f16(false, a, false, b2, (short)0, a2, false, false);
    a3 = __builtin_amdgcn_wmma_f32_16x16x32_f16(false, a, false, b3, (short)0, a3, false, false);
  }
  float* Hb = H + (size_t)b * FRED * NKG;
#pragma unroll
  for (int r = 0; r < 8; ++r) {
    const int m = mrow + r + 8 * hi;
    const float bs = bias[m];
    float* hm = Hb + (size_t)m * NKG;
    hm[p0]      = a0[r] + bs;
    hm[p0 + 16] = a1[r] + bs;
    hm[p0 + 32] = a2[r] + bs;
    hm[p0 + 48] = a3[r] + bs;
  }
}

// ---------------------------------------------------------------------------
// Train-mode BatchNorm stats per channel over (B, L): mean, rsqrt(var+eps)
// ---------------------------------------------------------------------------
__global__ void bn_stats_kernel(const float* __restrict__ Y, float* __restrict__ mean,
                                float* __restrict__ inv, int C, int L) {
  const int c = blockIdx.x;
  __shared__ float s1[256];
  __shared__ float s2[256];
  float sum = 0.f, sq = 0.f;
  const int total = BB * L;
  for (int i = threadIdx.x; i < total; i += blockDim.x) {
    const int b = i / L, j = i - b * L;
    const float v = Y[((size_t)b * C + c) * L + j];
    sum += v; sq += v * v;
  }
  s1[threadIdx.x] = sum; s2[threadIdx.x] = sq;
  __syncthreads();
  for (int s = 128; s > 0; s >>= 1) {
    if ((int)threadIdx.x < s) { s1[threadIdx.x] += s1[threadIdx.x + s]; s2[threadIdx.x] += s2[threadIdx.x + s]; }
    __syncthreads();
  }
  if (threadIdx.x == 0) {
    const float m = s1[0] / (float)total;
    const float v = s2[0] / (float)total - m * m;   // biased variance
    mean[c] = m;
    inv[c]  = rsqrtf(v + BN_EPS);
  }
}

// ---------------------------------------------------------------------------
// outT[(b*L+l)*C+c] = (half)( (add?add[i]:0) + relu(bn(Y[i])) ),  i over [B][C][L]
// (f16 output written transposed = next GEMM's [L][K] B-operand layout)
// ---------------------------------------------------------------------------
__global__ void bn_relu_add_f16t_kernel(const float* __restrict__ Y, const float* __restrict__ add,
                                        const float* __restrict__ mean, const float* __restrict__ inv,
                                        const float* __restrict__ g, const float* __restrict__ beta,
                                        half_t* __restrict__ outT, int C, int L) {
  const size_t i = (size_t)blockIdx.x * blockDim.x + threadIdx.x;
  if (i >= (size_t)BB * C * L) return;
  const int l = (int)(i % L);
  const int c = (int)((i / L) % C);
  const int b = (int)(i / ((size_t)C * L));
  float v = (Y[i] - mean[c]) * inv[c] * g[c] + beta[c];
  v = fmaxf(v, 0.f);
  if (add) v += add[i];
  outT[((size_t)b * L + l) * C + c] = (half_t)v;
}

// Final stage: canonical [B][C][L] f32 to d_out
__global__ void bn_relu_f32_kernel(const float* __restrict__ Y,
                                   const float* __restrict__ mean, const float* __restrict__ inv,
                                   const float* __restrict__ g, const float* __restrict__ beta,
                                   float* __restrict__ out, int C, int L) {
  const size_t i = (size_t)blockIdx.x * blockDim.x + threadIdx.x;
  if (i >= (size_t)BB * C * L) return;
  const int c = (int)((i / L) % C);
  const float v = (Y[i] - mean[c]) * inv[c] * g[c] + beta[c];
  out[i] = fmaxf(v, 0.f);
}

// ---------------------------------------------------------------------------
// Group epilogue: BN + ReLU + max-pool over KG.
// x2  (f32, canonical [B][C][N])  for laplacian gather + residual add
// x2t (f16, transposed [B][N][C]) for the next GEMM (not used directly here
//      but kept to feed the laplacian-transform operand path symmetrically)
// ---------------------------------------------------------------------------
__global__ void bn_relu_maxpool_kernel(const float* __restrict__ H,
                                       const float* __restrict__ mean, const float* __restrict__ inv,
                                       const float* __restrict__ g, const float* __restrict__ beta,
                                       float* __restrict__ x2, half_t* __restrict__ x2t) {
  const size_t i = (size_t)blockIdx.x * blockDim.x + threadIdx.x;   // over B*FRED*NPTS
  if (i >= (size_t)BB * FRED * NPTS) return;
  const int n = (int)(i % NPTS);
  const int c = (int)((i / NPTS) % FRED);
  const int b = (int)(i / ((size_t)FRED * NPTS));
  const size_t base = i * KG;                    // H canonical [B][C][N*KG]
  const float mc = mean[c], ic = inv[c], gc = g[c], bc = beta[c];
  float mx = -3.4e38f;
#pragma unroll
  for (int k = 0; k < KG; ++k) {
    float v = (H[base + k] - mc) * ic * gc + bc;
    mx = fmaxf(mx, fmaxf(v, 0.f));
  }
  x2[i] = mx;
  x2t[((size_t)b * NPTS + n) * FRED + c] = (half_t)mx;
}

// ---------------------------------------------------------------------------
// FeatureLaplacian with faithful [F,N*16] -> [N,16,F] reinterpret:
//   lf[n,kk,f] = x2[b, L>>15, idx2[b,(L&32767)>>4, L&15]], L = n*1024+kk*64+f
// Output stored transposed [B][N][F] f16 == natural order of i -> laph[i].
// ---------------------------------------------------------------------------
__global__ void laplacian_kernel(const float* __restrict__ x2, const int* __restrict__ idx16,
                                 half_t* __restrict__ lapT) {
  const size_t i = (size_t)blockIdx.x * blockDim.x + threadIdx.x;   // (b*N + n)*64 + f
  if (i >= (size_t)BB * NPTS * FRED) return;
  const int f = (int)(i & 63);
  const int n = (int)((i >> 6) & (NPTS - 1));
  const int b = (int)(i >> 17);
  const float* xb = x2 + (size_t)b * FRED * NPTS;
  const int* ib = idx16 + (size_t)b * NPTS * KLU;
  float s = 0.f;
#pragma unroll
  for (int kk = 0; kk < KLU; ++kk) {
    const int Lidx = n * 1024 + kk * 64 + f;
    const int fs = Lidx >> 15;
    const int q  = Lidx & 32767;
    const int sp = ib[(q >> 4) * KLU + (q & 15)];
    s += xb[(size_t)fs * NPTS + sp];
  }
  lapT[i] = (half_t)(xb[(size_t)f * NPTS + n] - s * (1.f / KLU));
}

// ---------------------------------------------------------------------------
extern "C" void kernel_launch(void* const* d_in, const int* in_sizes, int n_in,
                              void* d_out, int out_size, void* d_ws, size_t ws_size,
                              hipStream_t stream) {
  (void)in_sizes; (void)n_in; (void)out_size; (void)ws_size;
  const float* xyz  = (const float*)d_in[0];
  const float* feat = (const float*)d_in[1];
  const float* w1 = (const float*)d_in[2];  const float* b1  = (const float*)d_in[3];
  const float* g1 = (const float*)d_in[4];  const float* be1 = (const float*)d_in[5];
  const float* wf = (const float*)d_in[6];  const float* bf  = (const float*)d_in[7];
  const float* gg = (const float*)d_in[8];  const float* bg  = (const float*)d_in[9];
  const float* wl = (const float*)d_in[10]; const float* bl  = (const float*)d_in[11];
  const float* gl = (const float*)d_in[12]; const float* bel = (const float*)d_in[13];
  const float* w2 = (const float*)d_in[14]; const float* b2  = (const float*)d_in[15];
  const float* g2 = (const float*)d_in[16]; const float* be2 = (const float*)d_in[17];
  const float* w3 = (const float*)d_in[18]; const float* b3  = (const float*)d_in[19];
  const float* g3 = (const float*)d_in[20]; const float* be3 = (const float*)d_in[21];

  char* ws = (char*)d_ws;
  size_t off = 0;
  auto carve = [&](size_t bytes) -> void* {
    void* p = ws + off;
    off = (off + bytes + 255) & ~(size_t)255;
    return p;
  };

  int*    idx16 = (int*)   carve((size_t)BB * NPTS * KLU * 4);
  half_t* w1h   = (half_t*)carve((size_t)FRED * NFEA * 2);
  half_t* wfh   = (half_t*)carve((size_t)FRED * FRED * 2);
  half_t* wlh   = (half_t*)carve((size_t)FRED * FRED * 2);
  half_t* w2h   = (half_t*)carve((size_t)NFEA * FRED * 2);
  half_t* w3h   = (half_t*)carve((size_t)2 * NFEA * NFEA * 2);
  half_t* featT = (half_t*)carve((size_t)BB * NPTS * NFEA * 2);   // [B][N][128]
  float*  y1    = (float*) carve((size_t)BB * FRED * NPTS * 4);   // [B][64][N]
  half_t* x1t   = (half_t*)carve((size_t)BB * NPTS * FRED * 2);   // [B][N][64]
  float*  H     = (float*) carve((size_t)BB * FRED * NKG * 4);    // [B][64][N*KG]
  float*  x2    = (float*) carve((size_t)BB * FRED * NPTS * 4);   // [B][64][N]
  half_t* x2t   = (half_t*)carve((size_t)BB * NPTS * FRED * 2);   // [B][N][64]
  half_t* lapT  = (half_t*)carve((size_t)BB * NPTS * FRED * 2);   // [B][N][64]
  float*  t     = (float*) carve((size_t)BB * FRED * NPTS * 4);   // [B][64][N]
  half_t* x3t   = (half_t*)carve((size_t)BB * NPTS * FRED * 2);   // [B][N][64]
  float*  y2    = (float*) carve((size_t)BB * NFEA * NPTS * 4);   // [B][128][N]
  half_t* zt    = (half_t*)carve((size_t)BB * NPTS * NFEA * 2);   // [B][N][128]
  float*  y3    = (float*) carve((size_t)BB * 2 * NFEA * NPTS * 4);
  float*  meanb = (float*) carve(256 * 4);
  float*  invb  = (float*) carve(256 * 4);

  // ---- kNN (one top-16 pass; group stage uses the first 8) ----
  knn_kernel<<<dim3(NPTS / 256, BB), 256, 0, stream>>>(xyz, idx16);

  // ---- f16 operand preparation ----
  auto cast = [&](const float* src, half_t* dst, int n) {
    cast_f16_kernel<<<(n + 255) / 256, 256, 0, stream>>>(src, dst, n);
  };
  cast(w1, w1h, FRED * NFEA);
  cast(wf, wfh, FRED * FRED);
  cast(wl, wlh, FRED * FRED);
  cast(w2, w2h, NFEA * FRED);
  cast(w3, w3h, 2 * NFEA * NFEA);
  {
    size_t tot = (size_t)BB * NFEA * NPTS;
    cast_feat_t_kernel<<<(tot + 255) / 256, 256, 0, stream>>>(feat, featT);
  }

  const int EW = 256;

  // ---- mlp1: y1 = W1 @ feat + b1 ; BN ; ReLU -> x1t ----
  gemm_bias_f16<<<dim3(NPTS / 64, FRED / 16, BB), 32, 0, stream>>>(w1h, featT, b1, y1, FRED, NFEA, NPTS);
  bn_stats_kernel<<<FRED, 256, 0, stream>>>(y1, meanb, invb, FRED, NPTS);
  {
    size_t tot = (size_t)BB * FRED * NPTS;
    bn_relu_add_f16t_kernel<<<(tot + EW - 1) / EW, EW, 0, stream>>>(y1, nullptr, meanb, invb, g1, be1, x1t, FRED, NPTS);
  }

  // ---- DSgroupMLP: gather-GEMM, BN, ReLU, max-pool over k ----
  gemm_group_f16<<<dim3(NKG / 64, FRED / 16, BB), 32, 0, stream>>>(wfh, x1t, idx16, bf, H);
  bn_stats_kernel<<<FRED, 256, 0, stream>>>(H, meanb, invb, FRED, NKG);
  {
    size_t tot = (size_t)BB * FRED * NPTS;
    bn_relu_maxpool_kernel<<<(tot + EW - 1) / EW, EW, 0, stream>>>(H, meanb, invb, gg, bg, x2, x2t);
  }

  // ---- FeatureLaplacian: lapT -> t = Wl @ lap + bl ; BN ; ReLU ; x += t ----
  {
    size_t tot = (size_t)BB * FRED * NPTS;
    laplacian_kernel<<<(tot + EW - 1) / EW, EW, 0, stream>>>(x2, idx16, lapT);
  }
  gemm_bias_f16<<<dim3(NPTS / 64, FRED / 16, BB), 32, 0, stream>>>(wlh, lapT, bl, t, FRED, FRED, NPTS);
  bn_stats_kernel<<<FRED, 256, 0, stream>>>(t, meanb, invb, FRED, NPTS);
  {
    size_t tot = (size_t)BB * FRED * NPTS;
    bn_relu_add_f16t_kernel<<<(tot + EW - 1) / EW, EW, 0, stream>>>(t, x2, meanb, invb, gl, bel, x3t, FRED, NPTS);
  }

  // ---- mlp2: y2 = W2 @ x3 + b2 ; BN ; ReLU ; + feat -> zt ----
  gemm_bias_f16<<<dim3(NPTS / 64, NFEA / 16, BB), 32, 0, stream>>>(w2h, x3t, b2, y2, NFEA, FRED, NPTS);
  bn_stats_kernel<<<NFEA, 256, 0, stream>>>(y2, meanb, invb, NFEA, NPTS);
  {
    size_t tot = (size_t)BB * NFEA * NPTS;
    bn_relu_add_f16t_kernel<<<(tot + EW - 1) / EW, EW, 0, stream>>>(y2, feat, meanb, invb, g2, be2, zt, NFEA, NPTS);
  }

  // ---- mlp3: y3 = W3 @ z + b3 ; BN ; ReLU -> d_out ----
  gemm_bias_f16<<<dim3(NPTS / 64, (2 * NFEA) / 16, BB), 32, 0, stream>>>(w3h, zt, b3, y3, 2 * NFEA, NFEA, NPTS);
  bn_stats_kernel<<<2 * NFEA, 256, 0, stream>>>(y3, meanb, invb, 2 * NFEA, NPTS);
  {
    size_t tot = (size_t)BB * 2 * NFEA * NPTS;
    bn_relu_f32_kernel<<<(tot + EW - 1) / EW, EW, 0, stream>>>(y3, meanb, invb, g3, be3, (float*)d_out, 2 * NFEA, NPTS);
  }
}
